// CrossAttention_43241730736385
// MI455X (gfx1250) — matrix operand order repaired and checked by reading, not compiled, not run on hardware
//
#include <hip/hip_runtime.h>
#include <hip/hip_bf16.h>

// Cross-attention (torch MultiheadAttention) on gfx1250.
// Pipeline: f32->bf16 preconvert -> Q/K proj (WMMA, out^T orientation, x-slab
// async-staged in LDS) -> V proj (transposed store, same staging) -> flash
// attention (S^T trick, online softmax, async-to-LDS K/V) -> O proj.
// All matmuls: v_wmma_f32_16x16x32_bf16, fp32 accumulate.

#define DMODEL 1024
#define NHEADS 16
#define HDIM   64
#define BATCH  4
#define LQ     512
#define LKV    2048

#define XPITCH 40  // 64x32 x-slab row pitch (halfwords), padded vs 32
#define KPITCH 72  // 32x64 K-slab row pitch, padded vs 64
#define VPITCH 40  // 64x32 V-slab row pitch, padded vs 32

typedef __attribute__((ext_vector_type(16))) __bf16 v16bf;
typedef __attribute__((ext_vector_type(8)))  float  v8f;

union BF16x16 {
  v16bf v;
  __bf16 e[16];
  unsigned int u[8];
  uint4 q[2];
};

__device__ __forceinline__ unsigned short bfbits(float f) {
  union { __bf16 h; unsigned short s; } u;
  u.h = (__bf16)f;
  return u.s;
}

__device__ __forceinline__ v8f wmma_bf16(v16bf a, v16bf b, v8f c) {
  // D = A(16x32) * B(32x16) + C, fp32 accumulate
  return __builtin_amdgcn_wmma_f32_16x16x32_bf16(
      false, a, false, b, (short)0, c, false, false);
}

// 16 consecutive bf16 -> B-fragment (lane-half holds 16 consecutive K)
__device__ __forceinline__ v16bf load_16bf(const unsigned short* __restrict__ p) {
  BF16x16 B;
  B.q[0] = *(const uint4*)(p);
  B.q[1] = *(const uint4*)(p + 8);
  return B.v;
}

// A-fragment from bf16 row: chunks {kbase..+7, kbase+16..+23}
__device__ __forceinline__ v16bf load_a_bf16(const unsigned short* __restrict__ row, int kbase) {
  BF16x16 A;
  A.q[0] = *(const uint4*)(row + kbase);
  A.q[1] = *(const uint4*)(row + kbase + 16);
  return A.v;
}

// LDS byte offset of a pointer into a __shared__ array
__device__ __forceinline__ unsigned lds_off(const void* p) {
  return (unsigned)(unsigned long long)(p);  // flat LDS addr: offset in [31:0]
}

// async global -> LDS copy of 16 bytes (ASYNCcnt-tracked)
__device__ __forceinline__ void async_ld_b128(unsigned lds, const void* g) {
  asm volatile("global_load_async_to_lds_b128 %0, %1, off"
               :
               : "v"(lds), "v"((unsigned long long)(size_t)g)
               : "memory");
}
__device__ __forceinline__ void wait_async0() {
#if __has_builtin(__builtin_amdgcn_s_wait_asynccnt)
  __builtin_amdgcn_s_wait_asynccnt(0);
#else
  asm volatile("s_wait_asynccnt 0x0" ::: "memory");
#endif
}

// ---------------------------------------------------------------------------
// f32 -> bf16 elementwise (8 elems/thread, b128 stores)
// ---------------------------------------------------------------------------
__global__ __launch_bounds__(256) void cvt_bf16_kernel(
    const float* __restrict__ in, unsigned short* __restrict__ out, int n8) {
  int i = blockIdx.x * 256 + threadIdx.x;
  if (i >= n8) return;
  float4 a = ((const float4*)in)[2 * i];
  float4 b = ((const float4*)in)[2 * i + 1];
  union { __bf16 e[8]; uint4 q; } r;
  r.e[0] = (__bf16)a.x; r.e[1] = (__bf16)a.y;
  r.e[2] = (__bf16)a.z; r.e[3] = (__bf16)a.w;
  r.e[4] = (__bf16)b.x; r.e[5] = (__bf16)b.y;
  r.e[6] = (__bf16)b.z; r.e[7] = (__bf16)b.w;
  ((uint4*)out)[i] = r.q;
}

// ---------------------------------------------------------------------------
// GEMM, transposed orientation: out[m,n] = x[m,:].W[n,:] + bias[n]
// out^T = A(W 32x32, two 16-row tiles) x B(x 32x16 from LDS slab).
// Wave tile: 32(n) x 64(m) -> 8 WMMAs per 32-k step.
// The 64-row x slab is shared by all 8 waves of a block and async-staged
// into LDS (double buffered; 1 b128 per thread per k-step).
// Grid: (M/64)*4 blocks of 256; wave -> ntile (w&31), block -> m-strip.
// ---------------------------------------------------------------------------
template <bool OUTF32>
__global__ __launch_bounds__(256) void gemm_xt_kernel(
    const unsigned short* __restrict__ x, const unsigned short* __restrict__ W,
    const float* __restrict__ bias, void* __restrict__ outraw) {
  __shared__ unsigned short shX[2][64 * XPITCH];

  int t = threadIdx.x;
  int lane = t & 31;
  int lm = lane & 15, lh = lane >> 4;
  int ntile = (blockIdx.x * 8 + (t >> 5)) & 31;
  int n0 = ntile << 5;
  int m0 = (blockIdx.x >> 2) << 6;

  const unsigned short* wrow0 = W + (size_t)(n0 + lm) * DMODEL;
  const unsigned short* wrow1 = W + (size_t)(n0 + 16 + lm) * DMODEL;

  // stage coords: 64 rows x 32 cols, 16B per thread
  int xr = t >> 2, xc = (t & 3) << 3;
  const unsigned short* xsrc = x + (size_t)(m0 + xr) * DMODEL + xc;

  async_ld_b128(lds_off(&shX[0][xr * XPITCH + xc]), xsrc);

  v8f a00 = {}, a01 = {}, a02 = {}, a03 = {};
  v8f a10 = {}, a11 = {}, a12 = {}, a13 = {};
  int ka = lh << 3, kb = lh << 4;

  for (int ks = 0; ks < DMODEL; ks += 32) {
    int cur = (ks >> 5) & 1;
    wait_async0();
    __syncthreads();
    if (ks + 32 < DMODEL)
      async_ld_b128(lds_off(&shX[cur ^ 1][xr * XPITCH + xc]), xsrc + ks + 32);

    v16bf A0 = load_16bf(wrow0 + ks + kb);
    v16bf A1 = load_16bf(wrow1 + ks + kb);
    {
      v16bf B = load_16bf(&shX[cur][(0 + lm) * XPITCH + kb]);
      a00 = wmma_bf16(A0, B, a00);
      a10 = wmma_bf16(A1, B, a10);
    }
    {
      v16bf B = load_16bf(&shX[cur][(16 + lm) * XPITCH + kb]);
      a01 = wmma_bf16(A0, B, a01);
      a11 = wmma_bf16(A1, B, a11);
    }
    {
      v16bf B = load_16bf(&shX[cur][(32 + lm) * XPITCH + kb]);
      a02 = wmma_bf16(A0, B, a02);
      a12 = wmma_bf16(A1, B, a12);
    }
    {
      v16bf B = load_16bf(&shX[cur][(48 + lm) * XPITCH + kb]);
      a03 = wmma_bf16(A0, B, a03);
      a13 = wmma_bf16(A1, B, a13);
    }
    __syncthreads();
  }

  v8f accs[2][4] = {{a00, a01, a02, a03}, {a10, a11, a12, a13}};
#pragma unroll
  for (int nt = 0; nt < 2; ++nt) {
    int nb = n0 + nt * 16 + (lh << 3);  // 8 consecutive n per lane
    float4 bs0 = *(const float4*)(bias + nb);
    float4 bs1 = *(const float4*)(bias + nb + 4);
    float bb[8] = {bs0.x, bs0.y, bs0.z, bs0.w, bs1.x, bs1.y, bs1.z, bs1.w};
#pragma unroll
    for (int s = 0; s < 4; ++s) {
      int mrow = m0 + s * 16 + lm;
      float o[8];
#pragma unroll
      for (int r = 0; r < 8; ++r) o[r] = accs[nt][s][r] + bb[r];
      if (OUTF32) {
        float* out = (float*)outraw + (size_t)mrow * DMODEL + nb;
        float4 s0 = {o[0], o[1], o[2], o[3]};
        float4 s1 = {o[4], o[5], o[6], o[7]};
        *(float4*)(out) = s0;
        *(float4*)(out + 4) = s1;
      } else {
        union { __bf16 e[8]; uint4 q; } st;
#pragma unroll
        for (int r = 0; r < 8; ++r) st.e[r] = (__bf16)o[r];
        *(uint4*)((unsigned short*)outraw + (size_t)mrow * DMODEL + nb) = st.q;
      }
    }
  }
}

// ---------------------------------------------------------------------------
// V projection, normal orientation, TRANSPOSED store:
// Vt[((b*16+h)*64+d), j] (j contiguous). Wave tile 32(n) x 64(j); the 64-row
// kv slab is async-staged in LDS (read as A-fragments here).
// ---------------------------------------------------------------------------
__global__ __launch_bounds__(256) void gemm_v_kernel(
    const unsigned short* __restrict__ x, const unsigned short* __restrict__ W,
    const float* __restrict__ bias, unsigned short* __restrict__ Vt) {
  __shared__ unsigned short shX[2][64 * XPITCH];

  int t = threadIdx.x;
  int lane = t & 31;
  int lm = lane & 15, lh = lane >> 4;
  int ntile = (blockIdx.x * 8 + (t >> 5)) & 31;
  int n0 = ntile << 5;
  int m0 = (blockIdx.x >> 2) << 6;  // 64 kv rows

  const unsigned short* wrow0 = W + (size_t)(n0 + lm) * DMODEL;
  const unsigned short* wrow1 = W + (size_t)(n0 + 16 + lm) * DMODEL;

  int xr = t >> 2, xc = (t & 3) << 3;
  const unsigned short* xsrc = x + (size_t)(m0 + xr) * DMODEL + xc;

  async_ld_b128(lds_off(&shX[0][xr * XPITCH + xc]), xsrc);

  v8f a00 = {}, a01 = {}, a02 = {}, a03 = {};
  v8f a10 = {}, a11 = {}, a12 = {}, a13 = {};
  int ka = lh << 3, kb = lh << 4;

  for (int ks = 0; ks < DMODEL; ks += 32) {
    int cur = (ks >> 5) & 1;
    wait_async0();
    __syncthreads();
    if (ks + 32 < DMODEL)
      async_ld_b128(lds_off(&shX[cur ^ 1][xr * XPITCH + xc]), xsrc + ks + 32);

    v16bf B0 = load_a_bf16(wrow0, ks + ka);
    v16bf B1 = load_a_bf16(wrow1, ks + ka);
    {
      v16bf A = load_a_bf16(&shX[cur][(0 + lm) * XPITCH], ka);
      a00 = wmma_bf16(A, B0, a00);
      a10 = wmma_bf16(A, B1, a10);
    }
    {
      v16bf A = load_a_bf16(&shX[cur][(16 + lm) * XPITCH], ka);
      a01 = wmma_bf16(A, B0, a01);
      a11 = wmma_bf16(A, B1, a11);
    }
    {
      v16bf A = load_a_bf16(&shX[cur][(32 + lm) * XPITCH], ka);
      a02 = wmma_bf16(A, B0, a02);
      a12 = wmma_bf16(A, B1, a12);
    }
    {
      v16bf A = load_a_bf16(&shX[cur][(48 + lm) * XPITCH], ka);
      a03 = wmma_bf16(A, B0, a03);
      a13 = wmma_bf16(A, B1, a13);
    }
    __syncthreads();
  }

  int b = m0 / LKV;
  v8f accs[2][4] = {{a00, a01, a02, a03}, {a10, a11, a12, a13}};
#pragma unroll
  for (int nt = 0; nt < 2; ++nt) {
    int n = n0 + nt * 16 + lm;
    float bv_ = bias[n];
    int h = n >> 6, d = n & 63;
    unsigned short* vrow =
        Vt + (size_t)((b * NHEADS + h) * HDIM + d) * LKV + (m0 % LKV) + (lh << 3);
#pragma unroll
    for (int s = 0; s < 4; ++s) {
      union { __bf16 e[8]; uint4 q; } st;
#pragma unroll
      for (int r = 0; r < 8; ++r) st.e[r] = (__bf16)(accs[nt][s][r] + bv_);
      *(uint4*)(vrow + (size_t)s * 16) = st.q;  // j advances by 16 per sub-tile
    }
  }
}

// ---------------------------------------------------------------------------
// Flash attention. One wave per (b,h,16 q-rows); the 8 waves of a block share
// the same (b,h), so K/V chunks are staged once per block into LDS with
// async-to-LDS copies (double buffered, ASYNCcnt + barrier synchronized).
// S^T = K_chunk x Q^T puts the softmax row in the lane index, and its C-layout
// is bit-identical to the A-fragment of P for ctx += P x V.
// ---------------------------------------------------------------------------
__global__ __launch_bounds__(256) void attn_kernel(
    const unsigned short* __restrict__ Qh, const unsigned short* __restrict__ Kh,
    const unsigned short* __restrict__ Vt, unsigned short* __restrict__ Ctx) {
  __shared__ unsigned short shK[2][32 * KPITCH];  // [j-chunk 32][k 64]
  __shared__ unsigned short shV[2][64 * VPITCH];  // [d 64][j-chunk 32]

  int w = blockIdx.x * 8 + (threadIdx.x >> 5);  // 0..2047
  int lane = threadIdx.x & 31;
  int t = threadIdx.x;
  int lm = lane & 15, lh = lane >> 4;
  int b = w >> 9;
  int h = (w >> 5) & 15;
  int m0 = (w & 31) << 4;
  int halfbase = lh << 3;

  const unsigned short* qrow =
      Qh + (size_t)(b * LQ + m0 + lm) * DMODEL + h * HDIM;
  int kb = lh << 4;
  v16bf qb0 = load_a_bf16(qrow, halfbase);
  v16bf qb1 = load_a_bf16(qrow + 32, halfbase);

  const unsigned short* Kbase = Kh + (size_t)(b * LKV) * DMODEL + h * HDIM;
  const unsigned short* Vbase = Vt + (size_t)((b * NHEADS + h) * HDIM) * LKV;

  // per-thread stage coordinates (one b128 each into shK and shV per chunk)
  int krow = t >> 3, kcol = (t & 7) << 3;  // 32 rows x 64 cols
  int vrow = t >> 2, vcol = (t & 3) << 3;  // 64 rows x 32 cols

  // prefetch chunk 0 into buffer 0
  async_ld_b128(lds_off(&shK[0][krow * KPITCH + kcol]),
                Kbase + (size_t)krow * DMODEL + kcol);
  async_ld_b128(lds_off(&shV[0][vrow * VPITCH + vcol]),
                Vbase + (size_t)vrow * LKV + vcol);

  const float cscale = 0.18033688011112042f;  // (1/sqrt(64)) * log2(e)
  float mrow = -1.0e30f, lrow = 0.0f;
  v8f acc0 = {}, acc1 = {}, acc2 = {}, acc3 = {};

  for (int j0 = 0; j0 < LKV; j0 += 32) {
    int cur = (j0 >> 5) & 1;
    wait_async0();
    __syncthreads();  // staged chunk visible to all waves

    if (j0 + 32 < LKV) {  // prefetch next chunk into the other buffer
      int nxt = cur ^ 1;
      async_ld_b128(lds_off(&shK[nxt][krow * KPITCH + kcol]),
                    Kbase + (size_t)(j0 + 32 + krow) * DMODEL + kcol);
      async_ld_b128(lds_off(&shV[nxt][vrow * VPITCH + vcol]),
                    Vbase + (size_t)vrow * LKV + (j0 + 32) + vcol);
    }

    // --- scores: S^T tiles from LDS ---
    const unsigned short* k0r = &shK[cur][lm * KPITCH];
    const unsigned short* k1r = &shK[cur][(16 + lm) * KPITCH];
    int ka = lh << 3;
    v8f s0 = {}, s1 = {};
    s0 = wmma_bf16(load_a_bf16(k0r, ka), qb0, s0);
    s0 = wmma_bf16(load_a_bf16(k0r + 32, ka), qb1, s0);
    s1 = wmma_bf16(load_a_bf16(k1r, ka), qb0, s1);
    s1 = wmma_bf16(load_a_bf16(k1r + 32, ka), qb1, s1);

    // --- online softmax (row = lane&15; partner lane = lane^16) ---
    float t0[8], t1[8], tmax = -1.0e30f;
#pragma unroll
    for (int r = 0; r < 8; ++r) {
      t0[r] = s0[r] * cscale;
      t1[r] = s1[r] * cscale;
      tmax = fmaxf(tmax, fmaxf(t0[r], t1[r]));
    }
    tmax = fmaxf(tmax, __shfl_xor(tmax, 16));
    float mnew = fmaxf(mrow, tmax);
    float alpha = exp2f(mrow - mnew);
    float p0[8], p1[8], ps = 0.0f;
#pragma unroll
    for (int r = 0; r < 8; ++r) {
      p0[r] = exp2f(t0[r] - mnew);
      p1[r] = exp2f(t1[r] - mnew);
      ps += p0[r] + p1[r];
    }
    ps += __shfl_xor(ps, 16);
    lrow = lrow * alpha + ps;
    mrow = mnew;

    // P is already in A-fragment layout: tile0 -> e[0..7], tile1 -> e[8..15]
    BF16x16 P;
#pragma unroll
    for (int r = 0; r < 8; ++r) {
      P.e[r] = (__bf16)p0[r];
      P.e[8 + r] = (__bf16)p1[r];
    }

    // rescale ctx accumulators (row index on the VGPR axis -> shfl gather)
#pragma unroll
    for (int r = 0; r < 8; ++r) {
      float ar = __shfl(alpha, halfbase + r);
      acc0[r] *= ar; acc1[r] *= ar; acc2[r] *= ar; acc3[r] *= ar;
    }

    acc0 = wmma_bf16(P.v, load_a_bf16(&shV[cur][(0  + lm) * VPITCH], halfbase), acc0);
    acc1 = wmma_bf16(P.v, load_a_bf16(&shV[cur][(16 + lm) * VPITCH], halfbase), acc1);
    acc2 = wmma_bf16(P.v, load_a_bf16(&shV[cur][(32 + lm) * VPITCH], halfbase), acc2);
    acc3 = wmma_bf16(P.v, load_a_bf16(&shV[cur][(48 + lm) * VPITCH], halfbase), acc3);

    __syncthreads();  // all waves done reading before buffer is overwritten
  }

  float linv = 1.0f / lrow;
#pragma unroll
  for (int r = 0; r < 8; ++r) {
    float li = __shfl(linv, halfbase + r);
    size_t row = (size_t)(b * LQ + m0 + halfbase + r) * DMODEL + h * HDIM + lm;
    Ctx[row + 0]  = bfbits(acc0[r] * li);
    Ctx[row + 16] = bfbits(acc1[r] * li);
    Ctx[row + 32] = bfbits(acc2[r] * li);
    Ctx[row + 48] = bfbits(acc3[r] * li);
  }
}

// ---------------------------------------------------------------------------
extern "C" void kernel_launch(void* const* d_in, const int* in_sizes, int n_in,
                              void* d_out, int out_size, void* d_ws, size_t ws_size,
                              hipStream_t stream) {
  const float* q  = (const float*)d_in[0];
  const float* kv = (const float*)d_in[1];
  const float* Wq = (const float*)d_in[2];
  const float* bq = (const float*)d_in[3];
  const float* Wk = (const float*)d_in[4];
  const float* bk = (const float*)d_in[5];
  const float* Wv = (const float*)d_in[6];
  const float* bv = (const float*)d_in[7];
  const float* Wo = (const float*)d_in[8];
  const float* bo = (const float*)d_in[9];
  float* out = (float*)d_out;

  const size_t SZ_Q = (size_t)BATCH * LQ * DMODEL;    // 2M elems
  const size_t SZ_KV = (size_t)BATCH * LKV * DMODEL;  // 8M elems
  const size_t SZ_W = (size_t)DMODEL * DMODEL;        // 1M elems

  unsigned short* p = (unsigned short*)d_ws;
  unsigned short* qb  = p; p += SZ_Q;
  unsigned short* kvb = p; p += SZ_KV;
  unsigned short* Wqb = p; p += SZ_W;
  unsigned short* Wkb = p; p += SZ_W;
  unsigned short* Wvb = p; p += SZ_W;
  unsigned short* Wob = p; p += SZ_W;
  unsigned short* Qh  = p; p += SZ_Q;
  unsigned short* Kh  = p; p += SZ_KV;
  unsigned short* Vt  = p; p += SZ_KV;
  unsigned short* Ctx = p;

  // f32 -> bf16 preconversion (memory-bound)
  cvt_bf16_kernel<<<(int)(SZ_Q / 8 / 256), 256, 0, stream>>>(q, qb, (int)(SZ_Q / 8));
  cvt_bf16_kernel<<<(int)(SZ_KV / 8 / 256), 256, 0, stream>>>(kv, kvb, (int)(SZ_KV / 8));
  cvt_bf16_kernel<<<(int)(SZ_W / 8 / 256), 256, 0, stream>>>(Wq, Wqb, (int)(SZ_W / 8));
  cvt_bf16_kernel<<<(int)(SZ_W / 8 / 256), 256, 0, stream>>>(Wk, Wkb, (int)(SZ_W / 8));
  cvt_bf16_kernel<<<(int)(SZ_W / 8 / 256), 256, 0, stream>>>(Wv, Wvb, (int)(SZ_W / 8));
  cvt_bf16_kernel<<<(int)(SZ_W / 8 / 256), 256, 0, stream>>>(Wo, Wob, (int)(SZ_W / 8));

  // Q proj: M=2048 -> (2048/64)*4 = 128 blocks
  gemm_xt_kernel<false><<<128, 256, 0, stream>>>(qb, Wqb, bq, Qh);
  // K proj: M=8192 -> 512 blocks
  gemm_xt_kernel<false><<<512, 256, 0, stream>>>(kvb, Wkb, bk, Kh);
  // V proj (transposed store): 512 blocks
  gemm_v_kernel<<<512, 256, 0, stream>>>(kvb, Wvb, bv, Vt);
  // attention: 4*16*32 = 2048 waves -> 256 blocks
  attn_kernel<<<256, 256, 0, stream>>>(Qh, Kh, Vt, Ctx);
  // O proj (bf16 in, f32 out): M=2048 -> 128 blocks
  gemm_xt_kernel<true><<<128, 256, 0, stream>>>(Ctx, Wob, bo, out);
}
